// Mamba2ForCausalLM_3745211482488
// MI455X (gfx1250) — compile-verified, matrix operand order checked by reading, not run on hardware
//
#include <hip/hip_runtime.h>
#include <hip/hip_bf16.h>

// ---------------- problem constants ----------------
#define BB   2
#define LL   4096
#define DM   2048      // D_MODEL
#define DI   4096      // D_INNER
#define DSV  128       // D_STATE
#define NH   64        // NHEADS
#define HD   64        // HEADDIM
#define CH   64        // CHUNK
#define NC   (LL/CH)   // 64 chunks
#define CDIM (DI + 2*DSV)        // 4352
#define DIP  (2*DI + 2*DSV + NH) // 8512

typedef __attribute__((ext_vector_type(16))) __bf16 bf16x16;
typedef __attribute__((ext_vector_type(4)))  __bf16 bf16x4;
typedef __attribute__((ext_vector_type(8)))  float  f32x8;

__device__ inline f32x8 zero8() {
  f32x8 v;
#pragma unroll
  for (int i = 0; i < 8; ++i) v[i] = 0.f;
  return v;
}

__device__ inline f32x8 wmma_bf16(bf16x16 a, bf16x16 b, f32x8 c) {
  return __builtin_amdgcn_wmma_f32_16x16x32_bf16(false, a, false, b, (short)0, c, false, false);
}

// k index inside a 16x32 bf16 A/B fragment for VGPR-pair j (0..7) and lane half
__device__ inline int frag_k(int j, int half) {
  return ((j < 4) ? (2 * j) : (16 + 2 * (j - 4))) + 8 * half;
}

// Generic bf16 fragment gather from LDS (element (major,k) at
// lds[(major0+lane&15)*major_stride + k*k_stride]).
__device__ inline bf16x16 load_frag(const __bf16* lds, int major0, int major_stride,
                                    int k0, int k_stride, int lane) {
  int mj = major0 + (lane & 15);
  int half = lane >> 4;
  bf16x16 f;
#pragma unroll
  for (int j = 0; j < 8; ++j) {
    int k = k0 + frag_k(j, half);
    f[2 * j]     = lds[mj * major_stride + k * k_stride];
    f[2 * j + 1] = lds[mj * major_stride + (k + 1) * k_stride];
  }
  return f;
}

__device__ inline float siluf(float x) { return x / (1.f + __expf(-x)); }

// CDNA5 async global->LDS copy: 16 bytes per lane, tracked by ASYNCcnt.
__device__ inline void async_copy_b128(unsigned lds_addr, const void* gaddr) {
  asm volatile("global_load_async_to_lds_b128 %0, %1, off"
               :: "v"(lds_addr), "v"(gaddr) : "memory");
}

// ---------------- fp32 -> bf16 pack (vector) ----------------
__global__ __launch_bounds__(256) void pack_bf16(const float* __restrict__ src,
                                                 __bf16* __restrict__ dst,
                                                 size_t n4) {
  size_t i = (size_t)blockIdx.x * 256 + threadIdx.x;
  if (i >= n4) return;
  float4 v = ((const float4*)src)[i];
  bf16x4 w;
  w[0] = (__bf16)v.x; w[1] = (__bf16)v.y; w[2] = (__bf16)v.z; w[3] = (__bf16)v.w;
  ((bf16x4*)dst)[i] = w;
}

// ---------------- big NT GEMM: bf16 in, double-buffered async LDS, WMMA ----
// C[M,N] = A[M,K] x B[N,K]^T.  M%256==0, K%32==0; only N may have a tail.
#define LDT 48   // LDS row pitch in bf16 (96 B: multiple of 16 B for b128)
__global__ __launch_bounds__(256) void gemm_nt_bf16p(const __bf16* __restrict__ A,
                                                     const __bf16* __restrict__ B,
                                                     float* __restrict__ C,
                                                     int M, int N, int K) {
  __shared__ __attribute__((aligned(16))) __bf16 sA[2][256][LDT];
  __shared__ __attribute__((aligned(16))) __bf16 sB[2][128][LDT];
  const int tid  = threadIdx.x;
  const int lane = tid & 31;
  const int wave = tid >> 5;      // 0..7
  const int wr   = wave >> 1;     // 0..3 -> 64-row strip
  const int wc   = wave & 1;      // 0..1 -> 64-col strip
  const int bm0  = blockIdx.y * 256;
  const int bn0  = blockIdx.x * 128;

  f32x8 acc[4][4];
#pragma unroll
  for (int i = 0; i < 4; ++i)
#pragma unroll
    for (int j = 0; j < 4; ++j) acc[i][j] = zero8();

  auto stage = [&](int buf, int k0) {
    unsigned aBase = (unsigned)(uintptr_t)&sA[buf][0][0];
    unsigned bBase = (unsigned)(uintptr_t)&sB[buf][0][0];
    // A tile 256x32 bf16 = 1024 16B chunks (4/row)
#pragma unroll
    for (int it = 0; it < 4; ++it) {
      int e = tid + it * 256;
      int r = e >> 2, c16 = (e & 3) * 8;
      async_copy_b128(aBase + (unsigned)(r * LDT + c16) * 2u,
                      A + (size_t)(bm0 + r) * K + k0 + c16);
    }
    // B tile 128x32 bf16 = 512 chunks
#pragma unroll
    for (int it = 0; it < 2; ++it) {
      int e = tid + it * 256;
      int r = e >> 2, c16 = (e & 3) * 8;
      int gn = bn0 + r;
      gn = (gn < N) ? gn : (N - 1);          // clamp: tail cols never stored
      async_copy_b128(bBase + (unsigned)(r * LDT + c16) * 2u,
                      B + (size_t)gn * K + k0 + c16);
    }
  };

  const int nk = K >> 5;
  stage(0, 0);
  __builtin_amdgcn_s_wait_asynccnt(0);
  __syncthreads();

  for (int ks = 0; ks < nk; ++ks) {
    const int cur = ks & 1;
    if (ks + 1 < nk) stage(cur ^ 1, (ks + 1) << 5);   // prefetch next tile

    const __bf16* ab = &sA[cur][0][0];
    const __bf16* bb = &sB[cur][0][0];
    bf16x16 bfr[4];
#pragma unroll
    for (int j = 0; j < 4; ++j)
      bfr[j] = load_frag(bb, wc * 64 + j * 16, LDT, 0, 1, lane);
#pragma unroll
    for (int i = 0; i < 4; ++i) {
      bf16x16 afr = load_frag(ab, wr * 64 + i * 16, LDT, 0, 1, lane);
#pragma unroll
      for (int j = 0; j < 4; ++j) acc[i][j] = wmma_bf16(afr, bfr[j], acc[i][j]);
    }
    __builtin_amdgcn_s_wait_asynccnt(0);
    __syncthreads();
  }

  const int half = lane >> 4;
#pragma unroll
  for (int i = 0; i < 4; ++i) {
    int mb = bm0 + wr * 64 + i * 16 + 8 * half;
#pragma unroll
    for (int j = 0; j < 4; ++j) {
      int n = bn0 + wc * 64 + j * 16 + (lane & 15);
      if (n >= N) continue;
#pragma unroll
      for (int r = 0; r < 8; ++r)
        C[(size_t)(mb + r) * N + n] = acc[i][j][r];
    }
  }
}

// ---------------- causal depthwise conv(4) + bias + SiLU -> bf16 ----------------
__global__ __launch_bounds__(256) void conv_silu(const float* __restrict__ zx,
                                                 const float* __restrict__ cw,
                                                 const float* __restrict__ cb,
                                                 __bf16* __restrict__ xbcb) {
  size_t idx = (size_t)blockIdx.x * 256 + threadIdx.x;
  if (idx >= (size_t)BB * LL * CDIM) return;
  int ch = (int)(idx % CDIM);
  size_t bl = idx / CDIM;
  int l = (int)(bl % LL);
  int b = (int)(bl / LL);
  float acc = cb[ch];
#pragma unroll
  for (int k = 0; k < 4; ++k) {
    int ll = l - 3 + k;
    if (ll >= 0) acc += zx[((size_t)b * LL + ll) * DIP + DI + ch] * cw[ch * 4 + k];
  }
  xbcb[idx] = (__bf16)siluf(acc);
}

// ---------------- dt = softplus(raw + bias) ----------------
__global__ __launch_bounds__(256) void dt_softplus(const float* __restrict__ zx,
                                                   const float* __restrict__ dt_bias,
                                                   float* __restrict__ dtb) {
  size_t idx = (size_t)blockIdx.x * 256 + threadIdx.x;
  if (idx >= (size_t)BB * LL * NH) return;
  int h = (int)(idx % NH);
  float x = zx[(idx / NH) * DIP + (DIP - NH) + h] + dt_bias[h];
  dtb[idx] = (x > 20.f) ? x : log1pf(__expf(x));
}

// ---------------- xdt = x * dt (bf16, WMMA-ready) ----------------
__global__ __launch_bounds__(256) void make_xdt(const __bf16* __restrict__ xbcb,
                                                const float* __restrict__ dtb,
                                                __bf16* __restrict__ xdt) {
  size_t i4 = (size_t)blockIdx.x * 256 + threadIdx.x;
  if (i4 >= (size_t)BB * LL * DI / 4) return;
  size_t row = i4 / (DI / 4);
  int d4 = (int)(i4 % (DI / 4));
  float dt = dtb[row * NH + (d4 >> 4)];
  bf16x4 v = ((const bf16x4*)(xbcb + row * CDIM))[d4];
  bf16x4 w;
#pragma unroll
  for (int t = 0; t < 4; ++t) w[t] = (__bf16)((float)v[t] * dt);
  ((bf16x4*)xdt)[i4] = w;
}

// ---------------- per-chunk cumsum of dt * (-exp(A_log)) ----------------
__global__ __launch_bounds__(256) void chunk_cumsum(const float* __restrict__ dtb,
                                                    const float* __restrict__ A_log,
                                                    float* __restrict__ acs) {
  int g = blockIdx.x * 256 + threadIdx.x;    // over B*NH*NC
  if (g >= BB * NH * NC) return;
  int c = g % NC;
  int h = (g / NC) % NH;
  int b = g / (NC * NH);
  float negA = -__expf(A_log[h]);
  float run = 0.f;
  size_t obase = (((size_t)(b * NH + h)) * NC + c) * CH;
  for (int l = 0; l < CH; ++l) {
    run += dtb[((size_t)b * LL + c * CH + l) * NH + h] * negA;
    acs[obase + l] = run;
  }
}

// ---------------- per (b,c,h): Y_diag + chunk-local states ----------------
__global__ __launch_bounds__(128) void ssd_diag_states(const __bf16* __restrict__ xbcb,
                                                       const __bf16* __restrict__ xdt,
                                                       const float* __restrict__ acs,
                                                       float* __restrict__ y,
                                                       float* __restrict__ st) {
  const int h = blockIdx.x, c = blockIdx.y, b = blockIdx.z;
  __shared__ __attribute__((aligned(16))) __bf16 sX[64][72];    // xdt [l][p]
  __shared__ __attribute__((aligned(16))) __bf16 sBm[64][136];  // B   [l][n]
  __shared__ __attribute__((aligned(16))) __bf16 sCm[64][136];  // C   [l][n]
  __shared__ float  sG[64][72];    // C B^T [l][s]
  __shared__ float  sA[64];

  const int tid = threadIdx.x, lane = tid & 31, wave = tid >> 5, half = lane >> 4;
  const size_t rbase = (size_t)b * LL + c * CH;

  {
    unsigned sX0 = (unsigned)(uintptr_t)&sX[0][0];
    unsigned sB0 = (unsigned)(uintptr_t)&sBm[0][0];
    unsigned sC0 = (unsigned)(uintptr_t)&sCm[0][0];
    // xdt tile 64x64 bf16 = 512 chunks
#pragma unroll
    for (int it = 0; it < 4; ++it) {
      int e = tid + it * 128;
      int l = e >> 3, c16 = (e & 7) * 8;
      async_copy_b128(sX0 + (unsigned)(l * 72 + c16) * 2u,
                      xdt + (rbase + l) * DI + h * HD + c16);
    }
    // B/C tiles 64x128 bf16 = 1024 chunks each
#pragma unroll
    for (int it = 0; it < 8; ++it) {
      int e = tid + it * 128;
      int l = e >> 4, c16 = (e & 15) * 8;
      async_copy_b128(sB0 + (unsigned)(l * 136 + c16) * 2u,
                      xbcb + (rbase + l) * CDIM + DI + c16);
      async_copy_b128(sC0 + (unsigned)(l * 136 + c16) * 2u,
                      xbcb + (rbase + l) * CDIM + DI + DSV + c16);
    }
  }
  if (tid < 64) sA[tid] = acs[(((size_t)(b * NH + h)) * NC + c) * CH + tid];
  __builtin_amdgcn_s_wait_asynccnt(0);
  __syncthreads();

  // ---- G = C (64x128) x B^T (128x64) ----
  {
    f32x8 g[4];
#pragma unroll
    for (int j = 0; j < 4; ++j) g[j] = zero8();
    for (int kk = 0; kk < 128; kk += 32) {
      bf16x16 afr = load_frag(&sCm[0][0], wave * 16, 136, kk, 1, lane);
#pragma unroll
      for (int j = 0; j < 4; ++j) {
        bf16x16 bfr = load_frag(&sBm[0][0], j * 16, 136, kk, 1, lane);
        g[j] = wmma_bf16(afr, bfr, g[j]);
      }
    }
#pragma unroll
    for (int j = 0; j < 4; ++j) {
      int n = j * 16 + (lane & 15);
      int mb = wave * 16 + 8 * half;
#pragma unroll
      for (int r = 0; r < 8; ++r) sG[mb + r][n] = g[j][r];
    }
  }
  __syncthreads();

  // ---- Y_diag = (G o tril(exp(acs_l - acs_s))) x xdt ----
  {
    f32x8 ya[4];
#pragma unroll
    for (int j = 0; j < 4; ++j) ya[j] = zero8();
    const int m = wave * 16 + (lane & 15);
    const float am = sA[m];
    for (int kk = 0; kk < 64; kk += 32) {
      bf16x16 afr;
#pragma unroll
      for (int j8 = 0; j8 < 8; ++j8) {
        int k = kk + frag_k(j8, half);
        float v0 = (k     <= m) ? sG[m][k]     * __expf(am - sA[k])     : 0.f;
        float v1 = (k + 1 <= m) ? sG[m][k + 1] * __expf(am - sA[k + 1]) : 0.f;
        afr[2 * j8]     = (__bf16)v0;
        afr[2 * j8 + 1] = (__bf16)v1;
      }
#pragma unroll
      for (int j = 0; j < 4; ++j) {
        bf16x16 bfr = load_frag(&sX[0][0], j * 16, 1, kk, 72, lane); // [k=l][n=p]
        ya[j] = wmma_bf16(afr, bfr, ya[j]);
      }
    }
#pragma unroll
    for (int j = 0; j < 4; ++j) {
      int p = j * 16 + (lane & 15);
      int lb = wave * 16 + 8 * half;
#pragma unroll
      for (int r = 0; r < 8; ++r) {
        size_t row = rbase + lb + r;
        y[(row * NH + h) * HD + p] = ya[j][r];
      }
    }
  }

  // ---- chunk-local states[p][n] = sum_l xdt[l][p]*decay[l] * B[l][n] ----
  {
    f32x8 sa[8];
#pragma unroll
    for (int j = 0; j < 8; ++j) sa[j] = zero8();
    const float alast = sA[63];
    const int p = wave * 16 + (lane & 15);
    for (int kk = 0; kk < 64; kk += 32) {
      bf16x16 afr;
#pragma unroll
      for (int j8 = 0; j8 < 8; ++j8) {
        int k = kk + frag_k(j8, half);   // k = l
        afr[2 * j8]     = (__bf16)((float)sX[k][p]     * __expf(alast - sA[k]));
        afr[2 * j8 + 1] = (__bf16)((float)sX[k + 1][p] * __expf(alast - sA[k + 1]));
      }
#pragma unroll
      for (int j = 0; j < 8; ++j) {
        bf16x16 bfr = load_frag(&sBm[0][0], j * 16, 1, kk, 136, lane); // [k=l][n]
        sa[j] = wmma_bf16(afr, bfr, sa[j]);
      }
    }
    size_t base = (((size_t)b * NC + c) * NH + h) * (size_t)(HD * DSV);
#pragma unroll
    for (int j = 0; j < 8; ++j) {
      int n = j * 16 + (lane & 15);
      int pb = wave * 16 + 8 * half;
#pragma unroll
      for (int r = 0; r < 8; ++r) st[base + (size_t)(pb + r) * DSV + n] = sa[j][r];
    }
  }
}

// ---------------- sequential chunk scan: fp32 locals -> bf16 prefix ----------------
__global__ __launch_bounds__(256) void state_scan(const float* __restrict__ st,
                                                  __bf16* __restrict__ stp,
                                                  const float* __restrict__ acs) {
  const int h = blockIdx.x, b = blockIdx.y;
  const int tid = threadIdx.x;
  float run[32];
#pragma unroll
  for (int i = 0; i < 32; ++i) run[i] = 0.f;
  for (int c = 0; c < NC; ++c) {
    float dec = __expf(acs[(((size_t)(b * NH + h)) * NC + c) * CH + (CH - 1)]);
    size_t base = (((size_t)b * NC + c) * NH + h) * (size_t)(HD * DSV);
#pragma unroll
    for (int i = 0; i < 32; ++i) {
      size_t idx = base + tid + 256 * i;
      float local = st[idx];
      stp[idx] = (__bf16)run[i];        // prefix state entering chunk c
      run[i] = run[i] * dec + local;
    }
  }
}

// ---------------- Y_off + Y_diag + x*D, fused SiLU(z) gating ----------------
__global__ __launch_bounds__(128) void ssd_off_combine(const __bf16* __restrict__ xbcb,
                                                       const float* __restrict__ zx,
                                                       const float* __restrict__ acs,
                                                       const __bf16* __restrict__ stp,
                                                       const float* __restrict__ Dv,
                                                       float* __restrict__ y) {
  const int h = blockIdx.x, c = blockIdx.y, b = blockIdx.z;
  __shared__ __attribute__((aligned(16))) __bf16 sCm[64][136];  // C [l][n]
  __shared__ __attribute__((aligned(16))) __bf16 sS[64][136];   // prefix states [p][n]
  __shared__ float  sA[64];
  const int tid = threadIdx.x, lane = tid & 31, wave = tid >> 5, half = lane >> 4;
  const size_t rbase = (size_t)b * LL + c * CH;
  const size_t sbase = (((size_t)b * NC + c) * NH + h) * (size_t)(HD * DSV);

  {
    unsigned sC0 = (unsigned)(uintptr_t)&sCm[0][0];
    unsigned sS0 = (unsigned)(uintptr_t)&sS[0][0];
#pragma unroll
    for (int it = 0; it < 8; ++it) {
      int e = tid + it * 128;
      int l = e >> 4, c16 = (e & 15) * 8;
      async_copy_b128(sC0 + (unsigned)(l * 136 + c16) * 2u,
                      xbcb + (rbase + l) * CDIM + DI + DSV + c16);
      async_copy_b128(sS0 + (unsigned)(l * 136 + c16) * 2u,
                      stp + sbase + (size_t)l * DSV + c16);   // l plays p
    }
  }
  if (tid < 64) sA[tid] = acs[(((size_t)(b * NH + h)) * NC + c) * CH + tid];
  __builtin_amdgcn_s_wait_asynccnt(0);
  __syncthreads();

  f32x8 acc[4];
#pragma unroll
  for (int j = 0; j < 4; ++j) acc[j] = zero8();
  for (int kk = 0; kk < 128; kk += 32) {
    bf16x16 afr = load_frag(&sCm[0][0], wave * 16, 136, kk, 1, lane);
#pragma unroll
    for (int j = 0; j < 4; ++j) {
      bf16x16 bfr = load_frag(&sS[0][0], j * 16, 136, kk, 1, lane); // [p][n] NT
      acc[j] = wmma_bf16(afr, bfr, acc[j]);
    }
  }

  const float Dh = Dv[h];
#pragma unroll
  for (int j = 0; j < 4; ++j) {
    int p = j * 16 + (lane & 15);
    int lb = wave * 16 + 8 * half;
#pragma unroll
    for (int r = 0; r < 8; ++r) {
      int l = lb + r;
      size_t row = rbase + l;
      size_t yi = (row * NH + h) * HD + p;
      float yoff = acc[j][r] * __expf(sA[l]);
      float xv = (float)xbcb[row * CDIM + h * HD + p];
      float zv = zx[row * DIP + h * HD + p];
      y[yi] = (y[yi] + yoff + xv * Dh) * siluf(zv);
    }
  }
}

// ---------------- RMSNorm over D_INNER, bf16 output for out-GEMM ----------------
__global__ __launch_bounds__(256) void rmsnorm_rows(const float* __restrict__ y,
                                                    const float* __restrict__ nw,
                                                    __bf16* __restrict__ yo) {
  __shared__ float red[256];
  size_t base = (size_t)blockIdx.x * DI;
  float s = 0.f;
  for (int i = threadIdx.x; i < DI; i += 256) {
    float v = y[base + i];
    s += v * v;
  }
  red[threadIdx.x] = s;
  __syncthreads();
  for (int st = 128; st > 0; st >>= 1) {
    if (threadIdx.x < st) red[threadIdx.x] += red[threadIdx.x + st];
    __syncthreads();
  }
  float scale = rsqrtf(red[0] / (float)DI + 1e-5f);
  for (int i = threadIdx.x; i < DI; i += 256)
    yo[base + i] = (__bf16)(y[base + i] * scale * nw[i]);
}

// ---------------- host launch ----------------
extern "C" void kernel_launch(void* const* d_in, const int* in_sizes, int n_in,
                              void* d_out, int out_size, void* d_ws, size_t ws_size,
                              hipStream_t stream) {
  (void)in_sizes; (void)n_in; (void)out_size; (void)ws_size;
  const float* u       = (const float*)d_in[0];
  const float* W_in    = (const float*)d_in[1];
  const float* conv_w  = (const float*)d_in[2];
  const float* conv_b  = (const float*)d_in[3];
  const float* dt_bias = (const float*)d_in[4];
  const float* A_log   = (const float*)d_in[5];
  const float* Dv      = (const float*)d_in[6];
  const float* norm_w  = (const float*)d_in[7];
  const float* W_out   = (const float*)d_in[8];
  float* out = (float*)d_out;

  float* ws = (float*)d_ws;
  size_t off = 0;
  float* zx  = ws + off; off += (size_t)BB * LL * DIP;            // in_proj output
  float* dtb = ws + off; off += (size_t)BB * LL * NH;             // dt
  float* acs = ws + off; off += (size_t)BB * NH * NC * CH;        // cumsum(dA)
  float* yb  = ws + off; off += (size_t)BB * LL * DI;             // y (fp32)
  float* st  = ws + off; off += (size_t)BB * NC * NH * HD * DSV;  // chunk-local states
  __bf16* xbcb    = (__bf16*)(ws + off); off += ((size_t)BB * LL * CDIM) / 2;
  __bf16* xdt_bf  = (__bf16*)(ws + off); off += ((size_t)BB * LL * DI) / 2;
  __bf16* stp     = (__bf16*)(ws + off); off += ((size_t)BB * NC * NH * HD * DSV) / 2;
  __bf16* u_bf    = (__bf16*)(ws + off); off += ((size_t)BB * LL * DM) / 2;
  __bf16* win_bf  = (__bf16*)(ws + off); off += ((size_t)DIP * DM) / 2;
  __bf16* wout_bf = (__bf16*)(ws + off); off += ((size_t)DM * DI) / 2;
  __bf16* y_bf    = (__bf16*)(ws + off); off += ((size_t)BB * LL * DI) / 2;

  const int M = BB * LL;
  const size_t n4_u    = (size_t)M * DM / 4;
  const size_t n4_win  = (size_t)DIP * DM / 4;
  const size_t n4_wout = (size_t)DM * DI / 4;

  pack_bf16<<<(int)((n4_u + 255) / 256), dim3(256), 0, stream>>>(u, u_bf, n4_u);
  pack_bf16<<<(int)((n4_win + 255) / 256), dim3(256), 0, stream>>>(W_in, win_bf, n4_win);
  pack_bf16<<<(int)((n4_wout + 255) / 256), dim3(256), 0, stream>>>(W_out, wout_bf, n4_wout);

  gemm_nt_bf16p<<<dim3((DIP + 127) / 128, M / 256), dim3(256), 0, stream>>>(
      u_bf, win_bf, zx, M, DIP, DM);
  conv_silu<<<(int)(((size_t)M * CDIM + 255) / 256), dim3(256), 0, stream>>>(zx, conv_w, conv_b, xbcb);
  dt_softplus<<<(int)(((size_t)M * NH + 255) / 256), dim3(256), 0, stream>>>(zx, dt_bias, dtb);
  make_xdt<<<(int)(((size_t)M * DI / 4 + 255) / 256), dim3(256), 0, stream>>>(xbcb, dtb, xdt_bf);
  chunk_cumsum<<<(BB * NH * NC + 255) / 256, dim3(256), 0, stream>>>(dtb, A_log, acs);
  ssd_diag_states<<<dim3(NH, NC, BB), dim3(128), 0, stream>>>(xbcb, xdt_bf, acs, yb, st);
  state_scan<<<dim3(NH, BB), dim3(256), 0, stream>>>(st, stp, acs);
  ssd_off_combine<<<dim3(NH, NC, BB), dim3(128), 0, stream>>>(xbcb, zx, acs, stp, Dv, yb);
  rmsnorm_rows<<<M, dim3(256), 0, stream>>>(yb, norm_w, y_bf);
  gemm_nt_bf16p<<<dim3(DM / 128, M / 256), dim3(256), 0, stream>>>(
      y_bf, wout_bf, out, M, DM, DI);
}
